// MultiHeadAttention_34986803593394
// MI455X (gfx1250) — compile-verified, matrix operand order checked by reading, not run on hardware
//
#include <hip/hip_runtime.h>
#include <hip/hip_bf16.h>

// ---------------------------------------------------------------------------
// Transformer encoder block (MHA + residual + LayerNorm) for MI455X / gfx1250.
// All GEMMs on the WMMA pipe: v_wmma_f32_16x16x32_f16, f32 accumulate.
// wave32; 256-thread blocks = 8 waves. All LDS operand tiles are K-major so
// every fragment load is two ds_load_b128. f16 operand tiles are moved with
// GLOBAL_LOAD_ASYNC_TO_LDS_B128 (ASYNCcnt) to avoid the VGPR round-trip.
// ---------------------------------------------------------------------------

typedef __attribute__((ext_vector_type(16))) _Float16 v16h;
typedef __attribute__((ext_vector_type(8)))  _Float16 v8h;
typedef __attribute__((ext_vector_type(4)))  _Float16 v4h;
typedef __attribute__((ext_vector_type(8)))  float    v8f;
typedef __attribute__((ext_vector_type(4)))  float    v4f;

// Problem constants (match reference)
constexpr int Bc  = 2;
constexpr int Sc  = 2048;
constexpr int Dc  = 1024;
constexpr int Hc  = 16;
constexpr int DKc = 64;
constexpr int TDc = 3 * Dc;                  // 3072
constexpr float INV_TEMPER = 1.0f / 32.0f;   // 1/sqrt(1024)
constexpr float LN_EPS = 1e-6f;

constexpr int LDT = 48;   // LDS tile leading dim in halves (96B: keeps b128 alignment)

// ---------------------------------------------------------------------------
// WMMA fragment helper (ISA 7.12.2 layout, wave32). Tile is K-major in LDS.
// ---------------------------------------------------------------------------
__device__ __forceinline__ v16h load_frag_rows(const _Float16* base, int lane) {
    const _Float16* p = base + (lane & 15) * LDT + ((lane >> 4) << 3);
    v16h f;
#pragma unroll
    for (int i = 0; i < 8; ++i) f[i] = p[i];
#pragma unroll
    for (int i = 0; i < 8; ++i) f[8 + i] = p[16 + i];
    return f;
}

__device__ __forceinline__ v8f wmma_step(v16h a, v16h b, v8f c) {
    return __builtin_amdgcn_wmma_f32_16x16x32_f16(false, a, false, b, (short)0, c,
                                                  false, false);
}

// ---------------------------------------------------------------------------
// Async global(f16) -> LDS copy, 128 bits per lane per issue. The LDS target
// address is the low 32 bits of the generic pointer (LDS aperture keeps the
// byte offset in addr[31:0], ISA 10.2). Tracked by ASYNCcnt.
// ---------------------------------------------------------------------------
__device__ __forceinline__ void async_b128(_Float16* l, const _Float16* g) {
    unsigned la = (unsigned)(size_t)l;
    asm volatile("global_load_async_to_lds_b128 %0, %1, off"
                 :: "v"(la), "v"(g) : "memory");
}
__device__ __forceinline__ void wait_async() {
    asm volatile("s_wait_asynccnt 0x0" ::: "memory");
}

// ---------------------------------------------------------------------------
// Tile copy helpers (256 threads). All targets are K-major [rows][LDT] LDS.
// ---------------------------------------------------------------------------

// f32 global, row-major, K contiguous -> f16 LDS. 4 floats/op (b128 + cvt_pk).
template<int TM, int TK>
__device__ __forceinline__ void g2l_f32_rows(const float* __restrict__ g, int ld,
                                             _Float16* __restrict__ s, int tid) {
    constexpr int QPR = TK / 4;            // quads per row
#pragma unroll
    for (int e = tid; e < TM * QPR; e += 256) {
        int r = e / QPR, c4 = (e % QPR) * 4;
        v4f v = *(const v4f*)(g + (size_t)r * ld + c4);
        *(v4h*)(s + r * LDT + c4) = __builtin_convertvector(v, v4h);
    }
}

// f16 global, row-major, K contiguous -> f16 LDS via async-to-LDS b128.
template<int TM, int TK>
__device__ __forceinline__ void g2l_f16_rows_async(const _Float16* __restrict__ g, int ld,
                                                   _Float16* __restrict__ s, int tid) {
    constexpr int OPR = TK / 8;            // octets per row
#pragma unroll
    for (int e = tid; e < TM * OPR; e += 256) {
        int r = e / OPR, c8 = (e % OPR) * 8;
        async_b128(s + r * LDT + c8, g + (size_t)r * ld + c8);
    }
}

// f32 global tile [TK x TN] (N contiguous) -> f16 LDS transposed [TN][LDT].
template<int TK, int TN>
__device__ __forceinline__ void g2l_f32_trans(const float* __restrict__ g, int ld,
                                              _Float16* __restrict__ s, int tid) {
#pragma unroll
    for (int e = tid; e < (TK / 4) * TN; e += 256) {
        int n = e % TN, k4 = (e / TN) * 4;
        v4f v;
#pragma unroll
        for (int j = 0; j < 4; ++j) v[j] = g[(size_t)(k4 + j) * ld + n];
        *(v4h*)(s + n * LDT + k4) = __builtin_convertvector(v, v4h);
    }
}

// f16 global tile [TK x TN] (N contiguous) -> f16 LDS transposed [TN][LDT].
template<int TK, int TN>
__device__ __forceinline__ void g2l_f16_trans(const _Float16* __restrict__ g, int ld,
                                              _Float16* __restrict__ s, int tid) {
#pragma unroll
    for (int e = tid; e < (TK / 4) * TN; e += 256) {
        int n = e % TN, k4 = (e / TN) * 4;
        v4h v;
#pragma unroll
        for (int j = 0; j < 4; ++j) v[j] = g[(size_t)(k4 + j) * ld + n];
        *(v4h*)(s + n * LDT + k4) = v;
    }
}

// ---------------------------------------------------------------------------
// Generic NN GEMM: C[M,N] = epilogue(A[M,K] @ B[K,N]), B tile transposed in LDS.
// MODE 0: C = relu(acc + bias)            (QKV projection; TA=float, TO=f16)
// MODE 1: C = acc + bias + resid          (out projection;  TA=f16,  TO=float)
// BM = 128 (8 waves x 16 rows), BK = 32.
// ---------------------------------------------------------------------------
template<int BN, int MODE, typename TA, typename TO>
__global__ __launch_bounds__(256)
void gemm_nn(const TA* __restrict__ A, int lda,
             const float* __restrict__ Bm, int ldb,
             TO* __restrict__ C, int ldc, int K,
             const float* __restrict__ bias,
             const float* __restrict__ resid, int ldr) {
    constexpr int BM = 128, BK = 32, NT = BN / 16;
    __shared__ _Float16 sA[BM * LDT];
    __shared__ _Float16 sB[BN * LDT];    // transposed [N][K]

    const int tid  = threadIdx.x;
    const int lane = tid & 31;
    const int wave = tid >> 5;
    const int m0 = blockIdx.y * BM;
    const int n0 = blockIdx.x * BN;

    const TA*    Ag = A + (size_t)m0 * lda;
    const float* Bg = Bm + n0;

    v8f acc[NT];
#pragma unroll
    for (int t = 0; t < NT; ++t)
#pragma unroll
        for (int i = 0; i < 8; ++i) acc[t][i] = 0.0f;

    for (int kt = 0; kt < K; kt += BK) {
        if constexpr (sizeof(TA) == 2)
            g2l_f16_rows_async<BM, BK>((const _Float16*)Ag + kt, lda, sA, tid);
        else
            g2l_f32_rows<BM, BK>((const float*)Ag + kt, lda, sA, tid);
        g2l_f32_trans<BK, BN>(Bg + (size_t)kt * ldb, ldb, sB, tid);

        if (kt + BK < K) {   // hide next step's HBM latency (global_prefetch_b8)
            __builtin_prefetch(Ag + (size_t)(tid & (BM - 1)) * lda + kt + BK, 0, 1);
            __builtin_prefetch(Bg + (size_t)(kt + BK + (tid >> 5)) * ldb + lane * 4, 0, 1);
        }
        if constexpr (sizeof(TA) == 2) wait_async();
        __syncthreads();

        v16h af = load_frag_rows(sA + wave * 16 * LDT, lane);
#pragma unroll
        for (int t = 0; t < NT; ++t) {
            v16h bf = load_frag_rows(sB + t * 16 * LDT, lane);
            acc[t] = wmma_step(af, bf, acc[t]);
        }
        __syncthreads();
    }

    // C/D layout: lane half selects rows (+8), VGPR index r = row offset.
    const int rbase = m0 + wave * 16 + ((lane >> 4) << 3);
    const int cbase = n0 + (lane & 15);
#pragma unroll
    for (int t = 0; t < NT; ++t) {
        const int col = cbase + t * 16;
        const float bv = bias[col];
#pragma unroll
        for (int r = 0; r < 8; ++r) {
            const int row = rbase + r;
            float v = acc[t][r] + bv;
            if (MODE == 0) v = v > 0.0f ? v : 0.0f;
            if (MODE == 1) v += resid[(size_t)row * ldr + col];
            C[(size_t)row * ldc + col] = (TO)v;
        }
    }
}

// ---------------------------------------------------------------------------
// Attention scores (NT GEMM): scores[h*B+b, q, k] = (Qh[q,:].Kh[k,:]) / temper,
// masked. grid.z = h*B+b. K-dim = DK = 64: both k-step tiles are fetched as
// async-to-LDS copies up front (double buffers), one wait + one barrier, then
// all 16 WMMAs run with no intervening barrier.
// ---------------------------------------------------------------------------
__global__ __launch_bounds__(256)
void scores_nt(const _Float16* __restrict__ qkv, const unsigned char* __restrict__ mask,
               float* __restrict__ attn) {
    constexpr int BM = 128, BN = 128, BK = 32;
    constexpr int TILE = BM * LDT;
    __shared__ _Float16 sQ[2 * TILE];
    __shared__ _Float16 sK[2 * TILE];

    const int tid  = threadIdx.x;
    const int lane = tid & 31;
    const int wave = tid >> 5;
    const int z = blockIdx.z;
    const int b = z % Bc;
    const int h = z / Bc;
    const int q0 = blockIdx.y * BM;
    const int k0 = blockIdx.x * BN;

    const _Float16* Qg = qkv + (size_t)(b * Sc + q0) * TDc + h * DKc;        // qs
    const _Float16* Kg = qkv + (size_t)(b * Sc + k0) * TDc + Dc + h * DKc;   // ks
    float* Og = attn + (size_t)(h * Bc + b) * Sc * Sc;
    const unsigned char* Mg = mask + (size_t)b * Sc * Sc;

    v8f acc[8];
#pragma unroll
    for (int t = 0; t < 8; ++t)
#pragma unroll
        for (int i = 0; i < 8; ++i) acc[t][i] = 0.0f;

    // Issue all async tile copies (Q and K, both 32-wide K steps).
    g2l_f16_rows_async<BM, BK>(Qg,      TDc, sQ,        tid);
    g2l_f16_rows_async<BM, BK>(Qg + BK, TDc, sQ + TILE, tid);
    g2l_f16_rows_async<BN, BK>(Kg,      TDc, sK,        tid);
    g2l_f16_rows_async<BN, BK>(Kg + BK, TDc, sK + TILE, tid);
    wait_async();
    __syncthreads();

#pragma unroll
    for (int ks = 0; ks < 2; ++ks) {
        v16h af = load_frag_rows(sQ + ks * TILE + wave * 16 * LDT, lane);
#pragma unroll
        for (int t = 0; t < 8; ++t) {
            v16h bf = load_frag_rows(sK + ks * TILE + t * 16 * LDT, lane);
            acc[t] = wmma_step(af, bf, acc[t]);
        }
    }

    const int qb = q0 + wave * 16 + ((lane >> 4) << 3);
    const int kb = k0 + (lane & 15);
#pragma unroll
    for (int t = 0; t < 8; ++t) {
        const int kk = kb + t * 16;
#pragma unroll
        for (int r = 0; r < 8; ++r) {
            const int qq = qb + r;
            float v = acc[t][r] * INV_TEMPER;
            if (Mg[(size_t)qq * Sc + kk]) v = -__builtin_inff();
            Og[(size_t)qq * Sc + kk] = v;
        }
    }
}

// ---------------------------------------------------------------------------
// Row softmax over length S=2048, in place. One 256-thread block per row.
// ---------------------------------------------------------------------------
__global__ __launch_bounds__(256)
void softmax_rows(float* __restrict__ attn) {
    float* row = attn + (size_t)blockIdx.x * Sc;
    __shared__ float red[256];
    const int tid = threadIdx.x;

    float x[8];
    float mx = -__builtin_inff();
#pragma unroll
    for (int i = 0; i < 8; ++i) {
        x[i] = row[tid + i * 256];
        mx = fmaxf(mx, x[i]);
    }
    red[tid] = mx;
    __syncthreads();
    for (int s = 128; s > 0; s >>= 1) {
        if (tid < s) red[tid] = fmaxf(red[tid], red[tid + s]);
        __syncthreads();
    }
    mx = red[0];
    __syncthreads();

    float sum = 0.0f;
#pragma unroll
    for (int i = 0; i < 8; ++i) {
        x[i] = __expf(x[i] - mx);
        sum += x[i];
    }
    red[tid] = sum;
    __syncthreads();
    for (int s = 128; s > 0; s >>= 1) {
        if (tid < s) red[tid] += red[tid + s];
        __syncthreads();
    }
    const float inv = 1.0f / red[0];
#pragma unroll
    for (int i = 0; i < 8; ++i) row[tid + i * 256] = x[i] * inv;
}

// ---------------------------------------------------------------------------
// attn @ V per (b,h): [S x S]f32 @ [S x DK]f16 -> out[b, q, h*DK + :] (f16)
// ---------------------------------------------------------------------------
__global__ __launch_bounds__(256)
void attnv(const float* __restrict__ attn, const _Float16* __restrict__ qkv,
           _Float16* __restrict__ outp) {
    constexpr int BM = 128, BN = 64, BK = 32, NT = BN / 16;
    __shared__ _Float16 sA[BM * LDT];
    __shared__ _Float16 sB[BN * LDT];    // transposed [N][K]

    const int tid  = threadIdx.x;
    const int lane = tid & 31;
    const int wave = tid >> 5;
    const int z = blockIdx.z;
    const int b = z % Bc;
    const int h = z / Bc;
    const int q0 = blockIdx.y * BM;

    const float*    Ag = attn + (size_t)(h * Bc + b) * Sc * Sc + (size_t)q0 * Sc;
    const _Float16* Bg = qkv + (size_t)(b * Sc) * TDc + 2 * Dc + h * DKc;    // vs
    _Float16* Cg = outp + (size_t)(b * Sc + q0) * Dc + h * DKc;

    v8f acc[NT];
#pragma unroll
    for (int t = 0; t < NT; ++t)
#pragma unroll
        for (int i = 0; i < 8; ++i) acc[t][i] = 0.0f;

    for (int kt = 0; kt < Sc; kt += BK) {
        g2l_f32_rows<BM, BK>(Ag + kt, Sc, sA, tid);
        g2l_f16_trans<BK, BN>(Bg + (size_t)kt * TDc, TDc, sB, tid);

        if (kt + BK < Sc) {
            __builtin_prefetch(Ag + (size_t)(tid & (BM - 1)) * Sc + kt + BK, 0, 1);
            __builtin_prefetch(Bg + (size_t)(kt + BK + (tid >> 3)) * TDc + (tid & 7) * 8, 0, 1);
        }
        __syncthreads();

        v16h af = load_frag_rows(sA + wave * 16 * LDT, lane);
#pragma unroll
        for (int t = 0; t < NT; ++t) {
            v16h bf = load_frag_rows(sB + t * 16 * LDT, lane);
            acc[t] = wmma_step(af, bf, acc[t]);
        }
        __syncthreads();
    }

    const int rl = wave * 16 + ((lane >> 4) << 3);
    const int cl = lane & 15;
#pragma unroll
    for (int t = 0; t < NT; ++t)
#pragma unroll
        for (int r = 0; r < 8; ++r)
            Cg[(size_t)(rl + r) * Dc + cl + t * 16] = (_Float16)acc[t][r];
}

// ---------------------------------------------------------------------------
// LayerNorm over last dim (D=1024). One 256-thread block per row.
// ---------------------------------------------------------------------------
__global__ __launch_bounds__(256)
void layernorm_rows(const float* __restrict__ x, const float* __restrict__ gamma,
                    const float* __restrict__ beta, float* __restrict__ y) {
    const float* row = x + (size_t)blockIdx.x * Dc;
    float* yr = y + (size_t)blockIdx.x * Dc;
    __shared__ float rs[256];
    __shared__ float rq[256];
    const int tid = threadIdx.x;

    float v[4];
    float s = 0.0f, sq = 0.0f;
#pragma unroll
    for (int i = 0; i < 4; ++i) {
        v[i] = row[tid + i * 256];
        s += v[i];
        sq += v[i] * v[i];
    }
    rs[tid] = s;
    rq[tid] = sq;
    __syncthreads();
    for (int t = 128; t > 0; t >>= 1) {
        if (tid < t) { rs[tid] += rs[tid + t]; rq[tid] += rq[tid + t]; }
        __syncthreads();
    }
    const float mean = rs[0] * (1.0f / Dc);
    const float var  = rq[0] * (1.0f / Dc) - mean * mean;   // population var (jnp.var)
    const float rstd = rsqrtf(var + LN_EPS);
#pragma unroll
    for (int i = 0; i < 4; ++i) {
        const int c = tid + i * 256;
        yr[c] = (v[i] - mean) * rstd * gamma[c] + beta[c];
    }
}

// ---------------------------------------------------------------------------
// Launcher
// ---------------------------------------------------------------------------
extern "C" void kernel_launch(void* const* d_in, const int* in_sizes, int n_in,
                              void* d_out, int out_size, void* d_ws, size_t ws_size,
                              hipStream_t stream) {
    (void)in_sizes; (void)n_in; (void)out_size; (void)ws_size;

    const float* q      = (const float*)d_in[0];   // [B,S,D]  (k,v inputs unused)
    const unsigned char* mask = (const unsigned char*)d_in[3]; // [B,S,S] bool
    const float* W_qkv  = (const float*)d_in[4];   // [D, 3D]
    const float* b_qkv  = (const float*)d_in[5];   // [3D]
    const float* W_proj = (const float*)d_in[6];   // [D, D]
    const float* b_proj = (const float*)d_in[7];   // [D]
    const float* gamma  = (const float*)d_in[8];   // [D]
    const float* beta   = (const float*)d_in[9];   // [D]

    float* y_out    = (float*)d_out;                              // [B*S, D]
    float* attn_out = y_out + (size_t)Bc * Sc * Dc;               // [H*B, S, S]

    _Float16* qkvh = (_Float16*)d_ws;                             // [B*S, 3D] f16 (24MB)
    _Float16* attv = qkvh + (size_t)Bc * Sc * TDc;                // [B*S, D]  f16 (8MB)
    float*    xres = (float*)(attv + (size_t)Bc * Sc * Dc);       // [B*S, D]  f32 (16MB)

    const int M = Bc * Sc;                                        // 4096

    // 1) QKV = relu(X @ W_qkv + b_qkv) -> f16   [4096 x 3072 x 1024]
    gemm_nn<128, 0, float, _Float16><<<dim3(TDc / 128, M / 128), 256, 0, stream>>>(
        q, Dc, W_qkv, TDc, qkvh, TDc, Dc, b_qkv, nullptr, 0);

    // 2) scores = mask(Q.K^T / sqrt(D))  -> attn region of d_out (f32)
    scores_nt<<<dim3(Sc / 128, Sc / 128, Hc * Bc), 256, 0, stream>>>(
        qkvh, mask, attn_out);

    // 3) row softmax, in place
    softmax_rows<<<Hc * Bc * Sc, 256, 0, stream>>>(attn_out);

    // 4) attn @ V -> f16
    attnv<<<dim3(1, Sc / 128, Hc * Bc), 256, 0, stream>>>(attn_out, qkvh, attv);

    // 5) x = attv @ W_proj + b_proj + residual   [4096 x 1024 x 1024] -> f32
    gemm_nn<128, 1, _Float16, float><<<dim3(Dc / 128, M / 128), 256, 0, stream>>>(
        attv, Dc, W_proj, Dc, xres, Dc, Dc, b_proj, q, Dc);

    // 6) y = LayerNorm(x) * gamma + beta
    layernorm_rows<<<M, 256, 0, stream>>>(xres, gamma, beta, y_out);
}
